// EdgeUpdater_979252543696
// MI455X (gfx1250) — matrix operand order repaired and checked by reading, not compiled, not run on hardware
//
#include <hip/hip_runtime.h>

typedef __attribute__((ext_vector_type(16))) _Float16 v16h;
typedef __attribute__((ext_vector_type(8)))  _Float16 v8h;
typedef __attribute__((ext_vector_type(8)))  float    v8f;
typedef __attribute__((ext_vector_type(4)))  float    v4f;
typedef __attribute__((ext_vector_type(4)))  unsigned int v4u;

static __device__ __forceinline__ v16h combine16(v8h lo, v8h hi) {
  v16h r;
#pragma unroll
  for (int j = 0; j < 8; ++j) { r[j] = lo[j]; r[8 + j] = hi[j]; }
  return r;
}

// Pack two f32 into one dword holding (f16 lo, f16 hi) -> v_cvt_pk_f16_f32.
static __device__ __forceinline__ unsigned int pack2(float x0, float x1) {
  union { _Float16 h[2]; unsigned int u; } t;
  t.h[0] = (_Float16)x0;
  t.h[1] = (_Float16)x1;
  return t.u;
}

// De-interleave 8 pair-packed dwords (dword j = (a[j], a[8+j])) into the WMMA
// A-register layout using v_perm_b32 byte permutes (2 per dword pair).
static __device__ __forceinline__ v16h unpack_pairs(v4u q0, v4u q1) {
  unsigned qa[8] = {q0[0], q0[1], q0[2], q0[3], q1[0], q1[1], q1[2], q1[3]};
  union { unsigned u[8]; v16h v; } r;
#pragma unroll
  for (int t = 0; t < 4; ++t) {
    // A.dword[t]   = (lo(q_{2t}), lo(q_{2t+1}))
    r.u[t]     = __builtin_amdgcn_perm(qa[2 * t + 1], qa[2 * t], 0x05040100u);
    // A.dword[4+t] = (hi(q_{2t}), hi(q_{2t+1}))
    r.u[4 + t] = __builtin_amdgcn_perm(qa[2 * t + 1], qa[2 * t], 0x07060302u);
  }
  return r.v;
}

// A-matrix (16x32 f16) lane slice from a row-major f32 matrix row (32 cols),
// streamed with non-temporal hint. Lanes 0-15: K {0-7,16-23}; 16-31: {8-15,24-31}.
static __device__ __forceinline__ v16h loadA_f32_nt(const float* __restrict__ row, int lane) {
  const int kb = (lane & 16) ? 8 : 0;
  const float* p = row + kb;
  v4f x0 = __builtin_nontemporal_load((const v4f*)(p));
  v4f x1 = __builtin_nontemporal_load((const v4f*)(p + 4));
  v4f y0 = __builtin_nontemporal_load((const v4f*)(p + 16));
  v4f y1 = __builtin_nontemporal_load((const v4f*)(p + 20));
  v16h a;
#pragma unroll
  for (int j = 0; j < 4; ++j) {
    a[j]      = (_Float16)x0[j];
    a[4 + j]  = (_Float16)x1[j];
    a[8 + j]  = (_Float16)y0[j];
    a[12 + j] = (_Float16)y1[j];
  }
  return a;
}

// A-matrix lane slice from a row-major f16 row (gather tables: default temporal,
// L2-resident).
static __device__ __forceinline__ v16h loadA_f16(const _Float16* __restrict__ row, int lane) {
  const int kb = (lane & 16) ? 8 : 0;
  v8h lo = *(const v8h*)(row + kb);
  v8h hi = *(const v8h*)(row + kb + 16);
  return combine16(lo, hi);
}

// B-matrix (32x16 f16) lane slice for out = x @ W^T : B[k][n] = W[n][k].
static __device__ __forceinline__ v16h loadB(const _Float16* __restrict__ W, int Kdim,
                                             int nbase, int kchunk, int lane) {
  const int n  = nbase + (lane & 15);
  const int kb = kchunk * 32 + ((lane & 16) ? 8 : 0);
  const _Float16* p = W + (size_t)n * Kdim + kb;
  v8h lo = *(const v8h*)(p);
  v8h hi = *(const v8h*)(p + 16);
  return combine16(lo, hi);
}

static __device__ __forceinline__ v8f wmma_f16(v16h a, v16h b, v8f c) {
  return __builtin_amdgcn_wmma_f32_16x16x32_f16(false, a, false, b, (short)0, c, false, false);
}

// ---------------------------------------------------------------------------
// Weight fp32 -> fp16 conversion into workspace.
// layout (halfs): [0,1024) Wv1 [1024,2048) Wv2 [2048,3072) Wc1 [3072,4096) Wc2
//                 [4096,7168) We1 (32x96) [7168,8192) We2
// ---------------------------------------------------------------------------
__global__ void cvt_weights(const float* __restrict__ Wv1, const float* __restrict__ Wv2,
                            const float* __restrict__ Wc1, const float* __restrict__ Wc2,
                            const float* __restrict__ We1, const float* __restrict__ We2,
                            _Float16* __restrict__ dst) {
  int i = threadIdx.x + blockIdx.x * blockDim.x;
  if (i < 1024)      dst[i] = (_Float16)Wv1[i];
  else if (i < 2048) dst[i] = (_Float16)Wv2[i - 1024];
  else if (i < 3072) dst[i] = (_Float16)Wc1[i - 2048];
  else if (i < 4096) dst[i] = (_Float16)Wc2[i - 3072];
  else if (i < 7168) dst[i] = (_Float16)We1[i - 4096];
  else if (i < 8192) dst[i] = (_Float16)We2[i - 7168];
}

// ---------------------------------------------------------------------------
// Node compression MLP: dst_f16[N,32] = relu(relu(src@W1^T+b1)@W2^T+b2)
// One 16-row tile per wave; 4 WMMAs per tile.
// ---------------------------------------------------------------------------
__global__ void __launch_bounds__(256) node_mlp(
    const float* __restrict__ src, int nRows,
    const _Float16* __restrict__ W1, const float* __restrict__ b1,
    const _Float16* __restrict__ W2, const float* __restrict__ b2,
    _Float16* __restrict__ dst) {
  __shared__ __align__(16) unsigned int sU[8][256];  // reused: packed pairs / f16 tile
  const int lane = threadIdx.x & 31;
  const int wave = threadIdx.x >> 5;
  const int nTiles = (nRows + 15) >> 4;
  const int gwStride = gridDim.x * 8;

  const v16h B10 = loadB(W1, 32, 0, 0, lane);
  const v16h B11 = loadB(W1, 32, 16, 0, lane);
  const v16h B20 = loadB(W2, 32, 0, 0, lane);
  const v16h B21 = loadB(W2, 32, 16, 0, lane);
  const int n0 = lane & 15;
  const float b1lo = b1[n0], b1hi = b1[16 + n0];
  const float b2lo = b2[n0], b2hi = b2[16 + n0];
  const int moff = (lane & 16) ? 8 : 0;
  const int kb   = (lane & 16) ? 8 : 0;
  unsigned int* sp = sU[wave];
  _Float16* sh = (_Float16*)sU[wave];

  for (int tile = blockIdx.x * 8 + wave; tile < nTiles; tile += gwStride) {
    const int r0 = tile * 16;
    int rr = r0 + n0; if (rr > nRows - 1) rr = nRows - 1;
    v16h A = loadA_f32_nt(src + (size_t)rr * 32, lane);

    v8f c0 = {}, c1 = {};
    c0 = wmma_f16(A, B10, c0);
    c1 = wmma_f16(A, B11, c1);
    // bias + relu -> pair-packed LDS tile: dword p of row r = (col p, col p+16)
#pragma unroll
    for (int i = 0; i < 8; ++i) {
      float x0 = fmaxf(c0[i] + b1lo, 0.f);
      float x1 = fmaxf(c1[i] + b1hi, 0.f);
      sp[(i + moff) * 16 + n0] = pack2(x0, x1);
    }
    v4u q0 = *(const v4u*)&sp[n0 * 16 + kb];
    v4u q1 = *(const v4u*)&sp[n0 * 16 + kb + 4];
    v16h A2 = unpack_pairs(q0, q1);

    v8f d0 = {}, d1 = {};
    d0 = wmma_f16(A2, B20, d0);
    d1 = wmma_f16(A2, B21, d1);
    // bias + relu -> row-major f16 tile in LDS (for linear coalesced store)
#pragma unroll
    for (int i = 0; i < 8; ++i) {
      float x0 = fmaxf(d0[i] + b2lo, 0.f);
      float x1 = fmaxf(d1[i] + b2hi, 0.f);
      sh[(i + moff) * 32 + n0]      = (_Float16)x0;
      sh[(i + moff) * 32 + 16 + n0] = (_Float16)x1;
    }
    // coalesced f16 store of the 16x32 tile (512 halfs)
#pragma unroll
    for (int q = 0; q < 2; ++q) {
      int f = q * 256 + lane * 8;
      long g = (long)r0 * 32 + f;
      if (g < (long)nRows * 32) *(v8h*)(dst + g) = *(const v8h*)&sh[f];
    }
  }
}

// ---------------------------------------------------------------------------
// Edge MLP: out[E,32] = relu(concat(ef, v[iv], c[ic]) @ We1^T + be1) @ We2^T + be2
// One 16-edge tile per wave; 8 WMMAs per tile. Gather tables are f16 and
// L2-resident (12.8 MB << 192 MB L2); all streaming traffic is non-temporal.
// ---------------------------------------------------------------------------
__global__ void __launch_bounds__(256) edge_mlp(
    const float* __restrict__ ef, const int* __restrict__ eidx, int nEdges,
    const _Float16* __restrict__ vtab, const _Float16* __restrict__ ctab,
    const _Float16* __restrict__ W1, const float* __restrict__ b1,
    const _Float16* __restrict__ W2, const float* __restrict__ b2,
    float* __restrict__ out) {
  __shared__ __align__(16) unsigned int sHp[8][256];  // pair-packed hidden tile
  const int lane = threadIdx.x & 31;
  const int wave = threadIdx.x >> 5;
  const int nTiles = (nEdges + 15) >> 4;
  const int gwStride = gridDim.x * 8;

  // layer-1 B tiles: K=96 -> 3 chunks x 2 N-halves
  const v16h B1q0h0 = loadB(W1, 96, 0, 0, lane);
  const v16h B1q1h0 = loadB(W1, 96, 0, 1, lane);
  const v16h B1q2h0 = loadB(W1, 96, 0, 2, lane);
  const v16h B1q0h1 = loadB(W1, 96, 16, 0, lane);
  const v16h B1q1h1 = loadB(W1, 96, 16, 1, lane);
  const v16h B1q2h1 = loadB(W1, 96, 16, 2, lane);
  const v16h B20 = loadB(W2, 32, 0, 0, lane);
  const v16h B21 = loadB(W2, 32, 16, 0, lane);
  const int n0 = lane & 15;
  const float b1lo = b1[n0], b1hi = b1[16 + n0];
  const float b2lo = b2[n0], b2hi = b2[16 + n0];
  const int moff = (lane & 16) ? 8 : 0;
  const int kb   = (lane & 16) ? 8 : 0;
  unsigned int* sp = sHp[wave];

  for (int tile = blockIdx.x * 8 + wave; tile < nTiles; tile += gwStride) {
    const int e0 = tile * 16;
    int e = e0 + n0; if (e > nEdges - 1) e = nEdges - 1;
    const int iv = __builtin_nontemporal_load(eidx + e);
    const int ic = __builtin_nontemporal_load(eidx + nEdges + e);

    v16h Ae = loadA_f32_nt(ef + (size_t)e * 32, lane);   // K chunk 0 (stream)
    v16h Av = loadA_f16(vtab + (size_t)iv * 32, lane);   // K chunk 1 (L2 gather)
    v16h Ac = loadA_f16(ctab + (size_t)ic * 32, lane);   // K chunk 2 (L2 gather)

    v8f c0 = {}, c1 = {};
    c0 = wmma_f16(Ae, B1q0h0, c0);
    c0 = wmma_f16(Av, B1q1h0, c0);
    c0 = wmma_f16(Ac, B1q2h0, c0);
    c1 = wmma_f16(Ae, B1q0h1, c1);
    c1 = wmma_f16(Av, B1q1h1, c1);
    c1 = wmma_f16(Ac, B1q2h1, c1);

    // bias + relu -> pair-packed LDS tile (8x ds_store_b32 per lane)
#pragma unroll
    for (int i = 0; i < 8; ++i) {
      float x0 = fmaxf(c0[i] + b1lo, 0.f);
      float x1 = fmaxf(c1[i] + b1hi, 0.f);
      sp[(i + moff) * 16 + n0] = pack2(x0, x1);
    }
    v4u q0 = *(const v4u*)&sp[n0 * 16 + kb];
    v4u q1 = *(const v4u*)&sp[n0 * 16 + kb + 4];
    v16h A2 = unpack_pairs(q0, q1);

    v8f d0 = {}, d1 = {};
    d0 = wmma_f16(A2, B20, d0);
    d1 = wmma_f16(A2, B21, d1);

    // direct non-temporal stores from the D layout (64B segments -> full lines)
    float* orow = out + (size_t)(e0 + moff) * 32 + n0;
    if (e0 + 16 <= nEdges) {
#pragma unroll
      for (int i = 0; i < 8; ++i) {
        __builtin_nontemporal_store(d0[i] + b2lo, orow + (size_t)i * 32);
        __builtin_nontemporal_store(d1[i] + b2hi, orow + (size_t)i * 32 + 16);
      }
    } else {
#pragma unroll
      for (int i = 0; i < 8; ++i) {
        if (e0 + moff + i < nEdges) {
          __builtin_nontemporal_store(d0[i] + b2lo, orow + (size_t)i * 32);
          __builtin_nontemporal_store(d1[i] + b2hi, orow + (size_t)i * 32 + 16);
        }
      }
    }
  }
}

extern "C" void kernel_launch(void* const* d_in, const int* in_sizes, int n_in,
                              void* d_out, int out_size, void* d_ws, size_t ws_size,
                              hipStream_t stream) {
  const float* var_f  = (const float*)d_in[0];
  const float* con_f  = (const float*)d_in[1];
  const float* edge_f = (const float*)d_in[2];
  const int*   eidx   = (const int*)d_in[3];
  const float* Wv1 = (const float*)d_in[4];
  const float* bv1 = (const float*)d_in[5];
  const float* Wv2 = (const float*)d_in[6];
  const float* bv2 = (const float*)d_in[7];
  const float* Wc1 = (const float*)d_in[8];
  const float* bc1 = (const float*)d_in[9];
  const float* Wc2 = (const float*)d_in[10];
  const float* bc2 = (const float*)d_in[11];
  const float* We1 = (const float*)d_in[12];
  const float* be1 = (const float*)d_in[13];
  const float* We2 = (const float*)d_in[14];
  const float* be2 = (const float*)d_in[15];
  float* out = (float*)d_out;

  const int nVar  = in_sizes[0] / 32;
  const int nCon  = in_sizes[1] / 32;
  const int nEdge = in_sizes[2] / 32;

  _Float16* ws   = (_Float16*)d_ws;
  _Float16* vtab = ws;
  _Float16* ctab = vtab + (size_t)nVar * 32;
  _Float16* w16  = ctab + (size_t)nCon * 32;
  _Float16* Wv1h = w16;
  _Float16* Wv2h = w16 + 1024;
  _Float16* Wc1h = w16 + 2048;
  _Float16* Wc2h = w16 + 3072;
  _Float16* We1h = w16 + 4096;
  _Float16* We2h = w16 + 7168;

  cvt_weights<<<32, 256, 0, stream>>>(Wv1, Wv2, Wc1, Wc2, We1, We2, w16);

  const int varTiles  = (nVar + 15) / 16;
  const int conTiles  = (nCon + 15) / 16;
  const int edgeTiles = (nEdge + 15) / 16;

  dim3 blk(256);
  node_mlp<<<dim3((varTiles + 7) / 8), blk, 0, stream>>>(var_f, nVar, Wv1h, bv1, Wv2h, bv2, vtab);
  node_mlp<<<dim3((conTiles + 7) / 8), blk, 0, stream>>>(con_f, nCon, Wc1h, bc1, Wc2h, bc2, ctab);

  int edgeBlocks = (edgeTiles + 7) / 8;
  if (edgeBlocks > 4096) edgeBlocks = 4096;  // grid-stride loop amortizes weight loads
  edge_mlp<<<dim3(edgeBlocks), blk, 0, stream>>>(edge_f, eidx, nEdge, vtab, ctab,
                                                 We1h, be1, We2h, be2, out);
}
